// GeoMix3_33440615367379
// MI455X (gfx1250) — matrix operand order repaired and checked by reading, not compiled, not run on hardware
//
#include <hip/hip_runtime.h>
#include <hip/hip_bf16.h>

// ---------------------------------------------------------------------------
// 2-layer GCN: h = SpMM(A, x@W1+b1); BN+ReLU; out = SpMM(A, h@W2+b2)
// WMMA bf16 GEMMs: packed B staged in LDS per workgroup (one cooperative
// copy, then ds_load fragments feeding back-to-back WMMAs); A fragments
// loaded once per 16-row block and reused across all N tiles.
// SpMM scatter via hw float atomics; BN via LDS float atomics.
// ---------------------------------------------------------------------------

typedef __attribute__((ext_vector_type(16))) __bf16 v16bf;
typedef __attribute__((ext_vector_type(8)))  __bf16 v8bf;
typedef __attribute__((ext_vector_type(8)))  float  v8f;

__device__ __forceinline__ unsigned short f32_to_bf16(float f) {
    unsigned u = __builtin_bit_cast(unsigned, f);
    unsigned r = u + 0x7FFFu + ((u >> 16) & 1u);   // round-to-nearest-even
    return (unsigned short)(r >> 16);
}

// ------------------------------- utility -----------------------------------

__global__ void gcn_zero_f32(float* p, int n) {
    int i = blockIdx.x * blockDim.x + threadIdx.x;
    if (i < n) p[i] = 0.0f;
}

__global__ void gcn_cvt_bf16(const float* __restrict__ src,
                             unsigned short* __restrict__ dst, int n) {
    int i = blockIdx.x * blockDim.x + threadIdx.x;
    if (i < n) dst[i] = f32_to_bf16(src[i]);
}

// ------------------------- adjacency normalization --------------------------

__global__ void gcn_deg_accum(const long long* __restrict__ ei, float* __restrict__ deg,
                              int nEdges) {
    int e = blockIdx.x * blockDim.x + threadIdx.x;
    if (e >= nEdges) return;
    long long s = ei[e];
    long long d = ei[nEdges + e];
    if (s != d) unsafeAtomicAdd(&deg[d], 1.0f);
}

__global__ void gcn_dinv(float* __restrict__ degio, int n) {
    int i = blockIdx.x * blockDim.x + threadIdx.x;
    if (i < n) degio[i] = rsqrtf(degio[i] + 1.0f);   // deg incl. self loop >= 1
}

// ------------------------------ WMMA GEMM -----------------------------------
// Pack W (KxN row-major f32) into per-lane B fragments for
// v_wmma_f32_16x16x32_bf16:  lanes 0-15 hold K=0..15 (N = lane), lanes 16-31
// hold K=16..31 (N = lane-16); 16 contiguous halves per lane per (kt,nt) tile.
__global__ void gcn_pack_w(const float* __restrict__ W, unsigned short* __restrict__ Wp,
                           int N, int NT, int total) {
    int idx = blockIdx.x * blockDim.x + threadIdx.x;
    if (idx >= total) return;
    int h    = idx & 15;
    int lane = (idx >> 4) & 31;
    int tile = idx >> 9;
    int nt   = tile % NT;
    int kt   = tile / NT;
    int n = lane & 15;
    int k = kt * 32 + ((lane >> 4) << 4) + h;
    Wp[idx] = f32_to_bf16(W[(size_t)k * N + nt * 16 + n]);
}

// One wave computes a full 16 x N row-block.  The whole packed-B weight
// buffer (KT*NT*512 halves) is staged into LDS once per workgroup; inner
// loop reads B fragments via ds_load, so each WMMA depends only on a short
// LDS access instead of a VMEM round trip.
// A: M x K row-major bf16.  A-fragment per-lane k order used here (same as packed B):
//   lanes 0-15 (M=lane):   halves 0..15 = K 0..15
//   lanes 16-31 (M=l-16):  halves 0..15 = K 16..31
template <int NT, int KT>
__global__ void gcn_gemm_wmma(const unsigned short* __restrict__ Abf,
                              const unsigned short* __restrict__ Bpack,
                              const float* __restrict__ bias,
                              float* __restrict__ C,
                              int M) {
    const int N = NT * 16;
    const int K = KT * 32;
    const int MT = M >> 4;

    // ---- cooperative B stage (all threads; before any divergent exit) ----
    __shared__ unsigned short Bsh[KT * NT * 512];
    constexpr int CHUNKS = KT * NT * 64;          // 16-byte chunks
    for (int i = threadIdx.x; i < CHUNKS; i += blockDim.x)
        ((uint4*)Bsh)[i] = ((const uint4*)Bpack)[i];
    __syncthreads();

    int wave = (int)((blockIdx.x * blockDim.x + threadIdx.x) >> 5);
    if (wave >= MT) return;                       // wave-uniform guard
    int lane  = threadIdx.x & 31;
    int mA    = lane & 15;
    int koffA = (lane >> 4) << 4;
    const unsigned short* arow = Abf + (size_t)(wave * 16 + mA) * K;

    v8f acc[NT];
#pragma unroll
    for (int nt = 0; nt < NT; ++nt) acc[nt] = (v8f){};

#pragma unroll
    for (int kt = 0; kt < KT; ++kt) {
        int k0 = kt * 32;
        v8bf a0 = *(const v8bf*)(arow + k0 + koffA);
        v8bf a1 = *(const v8bf*)(arow + k0 + 8 + koffA);
        if (kt + 1 < KT)                          // pull next A tile toward WGP
            __builtin_prefetch(arow + k0 + 32, 0, 3);
        v16bf A;
#pragma unroll
        for (int i = 0; i < 8; ++i) { A[i] = a0[i]; A[i + 8] = a1[i]; }

#pragma unroll
        for (int nt = 0; nt < NT; ++nt) {
            const unsigned short* bp = Bsh + ((kt * NT + nt) * 32 + lane) * 16;
            v8bf b0 = *(const v8bf*)(bp);         // ds_load_b128
            v8bf b1 = *(const v8bf*)(bp + 8);
            v16bf B;
#pragma unroll
            for (int i = 0; i < 8; ++i) { B[i] = b0[i]; B[i + 8] = b1[i]; }
            acc[nt] = __builtin_amdgcn_wmma_f32_16x16x32_bf16(
                false, A, false, B, (short)0, acc[nt], false, false);
        }
    }
    int n     = lane & 15;
    int mBase = (lane >> 4) << 3;                 // D: vgpr r -> M = mBase + r
#pragma unroll
    for (int nt = 0; nt < NT; ++nt) {
        float bi = bias[nt * 16 + n];
#pragma unroll
        for (int r = 0; r < 8; ++r)
            C[(size_t)(wave * 16 + mBase + r) * N + nt * 16 + n] = acc[nt][r] + bi;
    }
}

// -------------------------------- SpMM --------------------------------------

__global__ void gcn_spmm_self(const float* __restrict__ dinv,
                              const float* __restrict__ h,
                              float* __restrict__ agg,
                              int logC, int total) {
    int i = blockIdx.x * blockDim.x + threadIdx.x;
    if (i >= total) return;
    float dv = dinv[i >> logC];
    agg[i] = dv * dv * h[i];
}

// One wave per edge: agg[src] += w * h[dst], coalesced over channels.
__global__ void gcn_spmm_edges(const long long* __restrict__ ei,
                               const float* __restrict__ dinv,
                               const float* __restrict__ h,
                               float* __restrict__ agg,
                               int C, int nEdges) {
    int e = (int)((blockIdx.x * blockDim.x + threadIdx.x) >> 5);
    if (e >= nEdges) return;
    long long s = ei[e];
    long long d = ei[nEdges + e];
    if (s == d) return;                           // wave-uniform
    float w = dinv[s] * dinv[d];
    const float* hrow = h + (size_t)d * C;
    float* arow = agg + (size_t)s * C;
    for (int c = threadIdx.x & 31; c < C; c += 32)
        unsafeAtomicAdd(&arow[c], w * hrow[c]);
}

// ------------------------------ BatchNorm -----------------------------------
// Per-channel sum / sum^2 with LDS float atomics, one global flush per block.
__global__ void gcn_bn_reduce(const float* __restrict__ h,
                              float* __restrict__ gsum, float* __restrict__ gsq,
                              int total) {
    __shared__ float ss[128];
    __shared__ float sq[128];
    if (threadIdx.x < 128) { ss[threadIdx.x] = 0.0f; sq[threadIdx.x] = 0.0f; }
    __syncthreads();
    for (int i = blockIdx.x * blockDim.x + threadIdx.x; i < total;
         i += gridDim.x * blockDim.x) {
        float v = h[i];
        int c = i & 127;
        atomicAdd(&ss[c], v);                     // ds_add_f32
        atomicAdd(&sq[c], v * v);
    }
    __syncthreads();
    if (threadIdx.x < 128) {
        unsafeAtomicAdd(&gsum[threadIdx.x], ss[threadIdx.x]);
        unsafeAtomicAdd(&gsq[threadIdx.x],  sq[threadIdx.x]);
    }
}

__global__ void gcn_bn_apply_relu_bf16(const float* __restrict__ h,
                                       const float* __restrict__ gsum,
                                       const float* __restrict__ gsq,
                                       const float* __restrict__ gamma,
                                       const float* __restrict__ beta,
                                       unsigned short* __restrict__ out,
                                       float invN, int total) {
    int i = blockIdx.x * blockDim.x + threadIdx.x;
    if (i >= total) return;
    int c = i & 127;
    float mean = gsum[c] * invN;
    float var  = gsq[c] * invN - mean * mean;
    float v = (h[i] - mean) * rsqrtf(var + 1e-5f) * gamma[c] + beta[c];
    out[i] = f32_to_bf16(v > 0.0f ? v : 0.0f);
}

// ------------------------------- launcher -----------------------------------

extern "C" void kernel_launch(void* const* d_in, const int* in_sizes, int n_in,
                              void* d_out, int out_size, void* d_ws, size_t ws_size,
                              hipStream_t stream) {
    const float*     x     = (const float*)d_in[0];
    const long long* ei    = (const long long*)d_in[1];
    const float*     W1    = (const float*)d_in[2];
    const float*     b1    = (const float*)d_in[3];
    const float*     gamma = (const float*)d_in[4];
    const float*     beta  = (const float*)d_in[5];
    const float*     W2    = (const float*)d_in[6];
    const float*     b2    = (const float*)d_in[7];
    float* out = (float*)d_out;

    const int IN_C = 128, HID_C = 128, OUT_C = 64;
    const int nNodes = in_sizes[0] / IN_C;        // 50000
    const int nEdges = in_sizes[1] / 2;           // 800000

    // --- workspace carve (hipMalloc is 256B-aligned) ---
    char* ws = (char*)d_ws;
    size_t off = 0;
    auto carve = [&](size_t bytes) { char* p = ws + off; off += (bytes + 255) & ~(size_t)255; return p; };
    float*          dinv  = (float*)carve((size_t)nNodes * 4);
    unsigned short* xb    = (unsigned short*)carve((size_t)nNodes * IN_C * 2); // reused as h1n
    unsigned short* w1p   = (unsigned short*)carve((size_t)(IN_C / 32) * (HID_C / 16) * 512 * 2);
    unsigned short* w2p   = (unsigned short*)carve((size_t)(HID_C / 32) * (OUT_C / 16) * 512 * 2);
    float*          h1    = (float*)carve((size_t)nNodes * HID_C * 4);          // reused as h2
    float*          h1agg = (float*)carve((size_t)nNodes * HID_C * 4);
    float*          bnsum = (float*)carve(HID_C * 4);
    float*          bnsq  = (float*)carve(HID_C * 4);
    unsigned short* h1n = xb;                     // reuse: xb dead after GEMM1
    float*          h2  = h1;                     // reuse: h1 dead after SpMM1

    const int T = 256;
    auto blocks = [](long long n, int t) { return (unsigned)((n + t - 1) / t); };

    // 1) degree -> dinv (deg accumulated in-place in `dinv` buffer)
    gcn_zero_f32<<<blocks(nNodes, T), T, 0, stream>>>(dinv, nNodes);
    gcn_deg_accum<<<blocks(nEdges, T), T, 0, stream>>>(ei, dinv, nEdges);
    gcn_dinv<<<blocks(nNodes, T), T, 0, stream>>>(dinv, nNodes);

    // 2) convert x to bf16; pack weights into WMMA B-fragment layout
    long long nx = (long long)nNodes * IN_C;
    gcn_cvt_bf16<<<blocks(nx, T), T, 0, stream>>>(x, xb, (int)nx);
    int w1tot = (IN_C / 32) * (HID_C / 16) * 512;
    int w2tot = (HID_C / 32) * (OUT_C / 16) * 512;
    gcn_pack_w<<<blocks(w1tot, T), T, 0, stream>>>(W1, w1p, HID_C, HID_C / 16, w1tot);
    gcn_pack_w<<<blocks(w2tot, T), T, 0, stream>>>(W2, w2p, OUT_C, OUT_C / 16, w2tot);

    // 3) GEMM1: h1 = xb @ W1 + b1   (one wave per 16-row block, 8 N tiles)
    long long waves1 = (long long)(nNodes / 16);
    gcn_gemm_wmma<8, 4><<<blocks(waves1 * 32, T), T, 0, stream>>>(xb, w1p, b1, h1,
                                                                  nNodes);

    // 4) SpMM1: h1agg = dinv^2*h1 + scatter(edges)
    long long nh1 = (long long)nNodes * HID_C;
    gcn_spmm_self<<<blocks(nh1, T), T, 0, stream>>>(dinv, h1, h1agg, 7, (int)nh1);
    gcn_spmm_edges<<<blocks((long long)nEdges * 32, T), T, 0, stream>>>(
        ei, dinv, h1, h1agg, HID_C, nEdges);

    // 5) BatchNorm (batch stats) + ReLU -> bf16
    gcn_zero_f32<<<1, 256, 0, stream>>>(bnsum, HID_C);
    gcn_zero_f32<<<1, 256, 0, stream>>>(bnsq, HID_C);
    gcn_bn_reduce<<<512, T, 0, stream>>>(h1agg, bnsum, bnsq, (int)nh1);
    gcn_bn_apply_relu_bf16<<<blocks(nh1, T), T, 0, stream>>>(
        h1agg, bnsum, bnsq, gamma, beta, h1n, 1.0f / (float)nNodes, (int)nh1);

    // 6) GEMM2: h2 = h1n @ W2 + b2   (one wave per 16-row block, 4 N tiles)
    gcn_gemm_wmma<4, 4><<<blocks(waves1 * 32, T), T, 0, stream>>>(h1n, w2p, b2, h2,
                                                                  nNodes);

    // 7) SpMM2 into d_out
    long long nh2 = (long long)nNodes * OUT_C;
    gcn_spmm_self<<<blocks(nh2, T), T, 0, stream>>>(dinv, h2, out, 6, (int)nh2);
    gcn_spmm_edges<<<blocks((long long)nEdges * 32, T), T, 0, stream>>>(
        ei, dinv, h2, out, OUT_C, nEdges);
}
